// Mamba3Mixer_21577915695226
// MI455X (gfx1250) — matrix operand order repaired and checked by reading, not compiled, not run on hardware
//
#include <hip/hip_runtime.h>
#include <hip/hip_bf16.h>
#include <cstdint>
#include <cstddef>

// ---------------------------------------------------------------------------
// Types for WMMA fragments (gfx1250 wave32)
// ---------------------------------------------------------------------------
typedef _Float16 v16h __attribute__((ext_vector_type(16)));
typedef _Float16 h8v  __attribute__((ext_vector_type(8)));
typedef _Float16 h4v  __attribute__((ext_vector_type(4)));
typedef float    v8f  __attribute__((ext_vector_type(8)));

// ---------------------------------------------------------------------------
// Problem constants (from setup_inputs)
// ---------------------------------------------------------------------------
constexpr int    Bb   = 2;
constexpr int    Ls   = 2048;
constexpr int    Dd   = 2048;     // D (model dim, GEMM K for projections)
constexpr int    DIN  = 4096;     // d_inner
constexpr int    Hh   = 64;       // heads
constexpr int    Nn   = 128;      // state dim
constexpr int    Pp   = 64;       // head dim
constexpr int    NH   = Hh * Nn;        // 8192
constexpr int    NH2  = Hh * (Nn / 2);  // 4096
constexpr size_t Mrows = (size_t)Bb * Ls; // 4096

// ---------------------------------------------------------------------------
// Workspace layout (bytes)
// ---------------------------------------------------------------------------
constexpr size_t o_xh    = 0;
constexpr size_t o_Wxh   = o_xh    + Mrows * Dd * 2;
constexpr size_t o_Wzh   = o_Wxh   + (size_t)DIN * Dd * 2;
constexpr size_t o_WBh   = o_Wzh   + (size_t)DIN * Dd * 2;
constexpr size_t o_WCh   = o_WBh   + (size_t)NH  * Dd * 2;
constexpr size_t o_Wdth  = o_WCh   + (size_t)NH  * Dd * 2;
constexpr size_t o_Wlamh = o_Wdth  + (size_t)Hh  * Dd * 2;
constexpr size_t o_Wthh  = o_Wlamh + (size_t)Hh  * Dd * 2;
constexpr size_t o_Wouth = o_Wthh  + (size_t)NH2 * Dd * 2;
constexpr size_t o_yh    = o_Wouth + (size_t)Dd  * DIN * 2;
constexpr size_t o_xp    = o_yh    + Mrows * DIN * 2;
constexpr size_t o_z     = o_xp    + Mrows * DIN * 4;
constexpr size_t o_Bm    = o_z     + Mrows * DIN * 4;
constexpr size_t o_Cm    = o_Bm    + Mrows * NH  * 4;
constexpr size_t o_th    = o_Cm    + Mrows * NH  * 4;
constexpr size_t o_dt    = o_th    + Mrows * NH2 * 4;
constexpr size_t o_lam   = o_dt    + Mrows * Hh  * 4;
constexpr size_t o_al    = o_lam   + Mrows * Hh  * 4;
constexpr size_t o_be    = o_al    + Mrows * Hh  * 4;
constexpr size_t o_ga    = o_be    + Mrows * Hh  * 4;
constexpr size_t o_y     = o_ga    + Mrows * Hh  * 4;

// ---------------------------------------------------------------------------
// fp32 -> fp16 conversion (vectorized 4-wide)
// ---------------------------------------------------------------------------
__global__ void mamba3_cvt_f16_kernel(const float* __restrict__ src,
                                      _Float16* __restrict__ dst, size_t n) {
  size_t i = ((size_t)blockIdx.x * blockDim.x + threadIdx.x) * 4;
  if (i < n) {
    float4 v = *(const float4*)(src + i);
    h4v o;
    o[0] = (_Float16)v.x; o[1] = (_Float16)v.y;
    o[2] = (_Float16)v.z; o[3] = (_Float16)v.w;
    *(h4v*)(dst + i) = o;
  }
}

// ---------------------------------------------------------------------------
// Async global->LDS copy helpers (CDNA5, tracked by ASYNCcnt)
// ---------------------------------------------------------------------------
__device__ __forceinline__ void async_ld_b128(uint32_t lds_addr,
                                              const _Float16* gptr) {
  asm volatile("global_load_async_to_lds_b128 %0, %1, off"
               :: "v"(lds_addr), "v"(gptr) : "memory");
}
__device__ __forceinline__ void wait_asynccnt0() {
  asm volatile("s_wait_asynccnt 0x0" ::: "memory");
}

// ---------------------------------------------------------------------------
// WMMA GEMM: C[M,Nc] = A[M,K](f16) @ W[Nc,K](f16)^T, fp32 accumulate.
// Block = 8 waves -> 128x64 tile. Wave = 32x32 (2x2 of 16x16 WMMA tiles).
// K-tiles (128x32 A, 64x32 W) are staged in double-buffered LDS via
// global_load_async_to_lds_b128; fragments re-read via ds_load_b128.
// Fragment layouts per CDNA5 ISA 7.12.2.
// ---------------------------------------------------------------------------
constexpr int AROW = 40;  // padded LDS row stride in halves (80B: 16B-aligned,
                          // rotates banks so 16-row fragment reads are conflict-free)

__global__ void __launch_bounds__(256)
mamba3_wmma_gemm_kernel(const _Float16* __restrict__ A,
                        const _Float16* __restrict__ W,
                        float* __restrict__ C,
                        int Kd, int Nc)
{
  __shared__ _Float16 As[2][128 * AROW];
  __shared__ _Float16 Wsm[2][64 * AROW];

  const int t    = threadIdx.x;
  const int lane = t & 31;
  const int wave = t >> 5;
  const int wm   = wave >> 1;               // 0..3
  const int wn   = wave & 1;                // 0..1
  const int bm0  = blockIdx.y * 128;
  const int bn0  = blockIdx.x * 64;
  const int l16  = lane & 15;
  const int hi   = lane >> 4;               // lane-half select

  // async copy assignment: chunk = (row, 16B part); thread t covers
  //   A rows (t>>2) and 64+(t>>2), W row (t>>2), part (t&3)
  const int crow  = t >> 2;                 // 0..63
  const int cpart = t & 3;                  // 0..3  (16B each, 64B per row)
  const uint32_t asb = (uint32_t)(uintptr_t)(&As[0][0]);
  const uint32_t wsb = (uint32_t)(uintptr_t)(&Wsm[0][0]);
  const uint32_t lofs = (uint32_t)(crow * AROW + cpart * 8) * 2;

  const _Float16* gA0 = A + (size_t)(bm0 + crow)      * Kd + cpart * 8;
  const _Float16* gA1 = A + (size_t)(bm0 + 64 + crow) * Kd + cpart * 8;
  const _Float16* gW  = W + (size_t)(bn0 + crow)      * Kd + cpart * 8;

  v8f acc00 = {}, acc01 = {}, acc10 = {}, acc11 = {};

  const int ra0 = wm * 32 + l16;
  const int rb0 = wn * 32 + l16;

  // prologue: stage first K-tile into buffer 0
  async_ld_b128(asb + lofs,                    gA0);
  async_ld_b128(asb + lofs + 64 * AROW * 2,    gA1);
  async_ld_b128(wsb + lofs,                    gW);

  int buf = 0;
  for (int kk = 0; kk < Kd; kk += 32) {
    wait_asynccnt0();       // own wave's async writes to LDS have landed
    __syncthreads();        // publish all waves' writes; retire prior reads

    if (kk + 32 < Kd) {     // stage next K-tile into the other buffer
      const int nb = buf ^ 1;
      async_ld_b128(asb + (uint32_t)nb * (128 * AROW * 2) + lofs,
                    gA0 + kk + 32);
      async_ld_b128(asb + (uint32_t)nb * (128 * AROW * 2) + lofs + 64 * AROW * 2,
                    gA1 + kk + 32);
      async_ld_b128(wsb + (uint32_t)nb * (64 * AROW * 2) + lofs,
                    gW + kk + 32);
    }

    const _Float16* Ab = &As[buf][0];
    const _Float16* Wb = &Wsm[buf][0];

    v16h a0, a1, b0, b1;
    // A fragment (16x32 f16): elems 0..7 -> K=hi*8+i, elems 8..15 -> K=16+hi*8+i
    {
      h8v lo = *(const h8v*)(Ab + ra0 * AROW + hi * 8);
      h8v up = *(const h8v*)(Ab + ra0 * AROW + 16 + hi * 8);
      #pragma unroll
      for (int i = 0; i < 8; i++) { a0[i] = lo[i]; a0[i + 8] = up[i]; }
    }
    {
      h8v lo = *(const h8v*)(Ab + (ra0 + 16) * AROW + hi * 8);
      h8v up = *(const h8v*)(Ab + (ra0 + 16) * AROW + 16 + hi * 8);
      #pragma unroll
      for (int i = 0; i < 8; i++) { a1[i] = lo[i]; a1[i + 8] = up[i]; }
    }
    // B fragment (32x16 f16): elems 0..7 -> K=hi*8+i, elems 8..15 -> K=16+hi*8+i, col = lane%16
    {
      h8v lo = *(const h8v*)(Wb + rb0 * AROW + hi * 8);
      h8v up = *(const h8v*)(Wb + rb0 * AROW + 16 + hi * 8);
      #pragma unroll
      for (int i = 0; i < 8; i++) { b0[i] = lo[i]; b0[i + 8] = up[i]; }
    }
    {
      h8v lo = *(const h8v*)(Wb + (rb0 + 16) * AROW + hi * 8);
      h8v up = *(const h8v*)(Wb + (rb0 + 16) * AROW + 16 + hi * 8);
      #pragma unroll
      for (int i = 0; i < 8; i++) { b1[i] = lo[i]; b1[i + 8] = up[i]; }
    }

    acc00 = __builtin_amdgcn_wmma_f32_16x16x32_f16(false, a0, false, b0, (short)0, acc00, false, false);
    acc01 = __builtin_amdgcn_wmma_f32_16x16x32_f16(false, a0, false, b1, (short)0, acc01, false, false);
    acc10 = __builtin_amdgcn_wmma_f32_16x16x32_f16(false, a1, false, b0, (short)0, acc10, false, false);
    acc11 = __builtin_amdgcn_wmma_f32_16x16x32_f16(false, a1, false, b1, (short)0, acc11, false, false);

    buf ^= 1;
  }

  // C/D layout: VGPR r -> M = r + 8*hi (lanes 0-15: M=r; 16-31: M=r+8), N = lane%16
  const int bm = bm0 + wm * 32;
  const int bn = bn0 + wn * 32;
  const int c0 = bn + l16;
  const int c1 = bn + 16 + l16;
  #pragma unroll
  for (int r = 0; r < 8; r++) {
    const int m0 = bm + r + 8 * hi;
    const int m1 = bm + 16 + r + 8 * hi;
    C[(size_t)m0 * Nc + c0] = acc00[r];
    C[(size_t)m0 * Nc + c1] = acc01[r];
    C[(size_t)m1 * Nc + c0] = acc10[r];
    C[(size_t)m1 * Nc + c1] = acc11[r];
  }
}

// ---------------------------------------------------------------------------
// rmsnorm over N=128 + bias, in place. One 128-thread block per (m,h).
// ---------------------------------------------------------------------------
__global__ void __launch_bounds__(128)
mamba3_rmsnorm_bias_kernel(float* __restrict__ buf,
                           const float* __restrict__ w,
                           const float* __restrict__ bias)
{
  const int grp = blockIdx.x;          // m*H + h
  const int h   = grp & (Hh - 1);
  const int n   = threadIdx.x;
  const size_t idx = (size_t)grp * Nn + n;
  const float v = buf[idx];
  float ss = v * v;
  #pragma unroll
  for (int o = 16; o > 0; o >>= 1) ss += __shfl_down(ss, o, 32);
  __shared__ float wsum[4];
  const int lane = threadIdx.x & 31, wv = threadIdx.x >> 5;
  if (lane == 0) wsum[wv] = ss;
  __syncthreads();
  const float tot = wsum[0] + wsum[1] + wsum[2] + wsum[3];
  const float scale = rsqrtf(tot * (1.0f / Nn) + 1e-6f);
  buf[idx] = v * scale * w[n] + bias[h * Nn + n];
}

// ---------------------------------------------------------------------------
// dt=softplus(dt_raw); lam=sigmoid(lam_raw); alpha/beta/gamma
// ---------------------------------------------------------------------------
__global__ void mamba3_dtlam_kernel(float* __restrict__ dt,
                                    const float* __restrict__ lam_raw,
                                    const float* __restrict__ A_log,
                                    float* __restrict__ alpha,
                                    float* __restrict__ beta,
                                    float* __restrict__ gamma,
                                    int total)
{
  const int i = blockIdx.x * blockDim.x + threadIdx.x;
  if (i >= total) return;
  const int h = i & (Hh - 1);
  const float xr = dt[i];
  const float sp = (xr > 20.f) ? xr : log1pf(expf(xr));
  const float lm = 1.f / (1.f + expf(-lam_raw[i]));
  const float A  = -expf(A_log[h]);
  const float al = expf(sp * A);
  dt[i]    = sp;
  alpha[i] = al;
  beta[i]  = (1.f - lm) * sp * al;
  gamma[i] = lm * sp;
}

// ---------------------------------------------------------------------------
// theta_cumsum[l] = cumsum_l( theta * dt ), in place. One thread per (b,h,j).
// ---------------------------------------------------------------------------
__global__ void mamba3_cumsum_kernel(float* __restrict__ theta,
                                     const float* __restrict__ dt)
{
  const int idx = blockIdx.x * blockDim.x + threadIdx.x;  // B*H*64 = 8192
  const int j  = idx & 63;
  const int bh = idx >> 6;
  const int h  = bh & (Hh - 1);
  const int b  = bh >> 6;
  float run = 0.f;
  #pragma unroll 1
  for (int l = 0; l < Ls; l++) {
    const size_t m = (size_t)b * Ls + l;
    run += theta[m * NH2 + h * 64 + j] * dt[m * Hh + h];
    theta[m * NH2 + h * 64 + j] = run;
  }
}

// ---------------------------------------------------------------------------
// data-dependent RoPE applied to Bm and Cm (pairs 2j,2j+1), in place
// ---------------------------------------------------------------------------
__global__ void mamba3_rope_kernel(float* __restrict__ Bm,
                                   float* __restrict__ Cm,
                                   const float* __restrict__ th)
{
  const size_t i = (size_t)blockIdx.x * blockDim.x + threadIdx.x; // M*H*64
  const size_t j  = i & 63;
  const size_t mh = i >> 6;
  const size_t h  = mh & (Hh - 1);
  const size_t m  = mh >> 6;
  const float t = th[i];                 // theta layout (M, H*64) == flat i
  float c, s;
  __sincosf(t, &s, &c);
  const size_t base = m * NH + h * Nn + 2 * j;
  const float b0 = Bm[base], b1 = Bm[base + 1];
  Bm[base]     = b0 * c - b1 * s;
  Bm[base + 1] = b0 * s + b1 * c;
  const float c0 = Cm[base], c1 = Cm[base + 1];
  Cm[base]     = c0 * c - c1 * s;
  Cm[base + 1] = c0 * s + c1 * c;
}

// ---------------------------------------------------------------------------
// Trapezoidal selective scan. One 256-thread block per (b,h); state h[128][64]
// fp32 entirely in VGPRs (32 per thread). B/C/x tiles staged in LDS each step.
// ---------------------------------------------------------------------------
__global__ void __launch_bounds__(256)
mamba3_scan_kernel(const float* __restrict__ Bm, const float* __restrict__ Cm,
                   const float* __restrict__ xp, const float* __restrict__ alpha,
                   const float* __restrict__ beta, const float* __restrict__ gamma,
                   float* __restrict__ y)
{
  const int bh = blockIdx.x;           // b*H + h
  const int b  = bh >> 6;
  const int h  = bh & (Hh - 1);
  const int t  = threadIdx.x;
  const int p  = t & 63;
  const int g  = t >> 6;               // n-group 0..3

  __shared__ float Bbuf[2][Nn];
  __shared__ float Cbuf[Nn];
  __shared__ float xbuf[Pp];
  __shared__ float ypart[256];

  float hst[32];
  #pragma unroll
  for (int i = 0; i < 32; i++) hst[i] = 0.f;
  float x_prev = 0.f;
  if (t < Nn) Bbuf[1][t] = 0.f;        // B_prev = 0 at l=0

  #pragma unroll 1
  for (int l = 0; l < Ls; l++) {
    const int cur = l & 1, prv = cur ^ 1;
    const size_t m = (size_t)b * Ls + l;
    const size_t bc = m * NH + (size_t)h * Nn;
    if (t < Nn) Bbuf[cur][t] = Bm[bc + t];
    else        Cbuf[t - Nn] = Cm[bc + (t - Nn)];
    if (t < Pp) xbuf[t] = xp[m * DIN + (size_t)h * Pp + t];
    __syncthreads();

    const size_t sidx = m * Hh + h;
    const float a  = alpha[sidx];
    const float bt = beta[sidx];
    const float gt = gamma[sidx];
    const float xc = xbuf[p];
    float acc = 0.f;
    #pragma unroll
    for (int i = 0; i < 32; i++) {
      const int n = g * 32 + i;
      const float hv = a * hst[i] + bt * Bbuf[prv][n] * x_prev
                                  + gt * Bbuf[cur][n] * xc;
      hst[i] = hv;
      acc += Cbuf[n] * hv;
    }
    x_prev = xc;
    ypart[t] = acc;
    __syncthreads();
    if (t < Pp)
      y[m * DIN + (size_t)h * Pp + t] =
          ypart[t] + ypart[64 + t] + ypart[128 + t] + ypart[192 + t];
  }
}

// ---------------------------------------------------------------------------
// y * silu(z), fused fp16 conversion for final GEMM
// ---------------------------------------------------------------------------
__global__ void mamba3_gate_cvt_kernel(const float* __restrict__ y,
                                       const float* __restrict__ z,
                                       _Float16* __restrict__ yh)
{
  const size_t i = (size_t)blockIdx.x * blockDim.x + threadIdx.x; // M*DIN
  const float zv = z[i];
  const float sv = zv / (1.f + expf(-zv));
  yh[i] = (_Float16)(y[i] * sv);
}

// ---------------------------------------------------------------------------
// Host launcher
// ---------------------------------------------------------------------------
static inline void launch_gemm(const _Float16* A, const _Float16* W, float* C,
                               int Kd, int Nc, hipStream_t stream) {
  dim3 grid(Nc / 64, (unsigned)(Mrows / 128));
  mamba3_wmma_gemm_kernel<<<grid, 256, 0, stream>>>(A, W, C, Kd, Nc);
}

static inline void launch_cvt(const float* s, _Float16* d, size_t n,
                              hipStream_t stream) {
  const size_t vec = n / 4;
  const int blocks = (int)((vec + 255) / 256);
  mamba3_cvt_f16_kernel<<<blocks, 256, 0, stream>>>(s, d, n);
}

extern "C" void kernel_launch(void* const* d_in, const int* in_sizes, int n_in,
                              void* d_out, int out_size, void* d_ws, size_t ws_size,
                              hipStream_t stream) {
  (void)in_sizes; (void)n_in; (void)out_size; (void)ws_size;

  const float* x     = (const float*)d_in[0];
  const float* Wx    = (const float*)d_in[1];
  const float* Wz    = (const float*)d_in[2];
  const float* WB    = (const float*)d_in[3];
  const float* WC    = (const float*)d_in[4];
  const float* Wdt   = (const float*)d_in[5];
  const float* Wlam  = (const float*)d_in[6];
  const float* Wth   = (const float*)d_in[7];
  const float* A_log = (const float*)d_in[8];
  const float* Bn_w  = (const float*)d_in[9];
  const float* Cn_w  = (const float*)d_in[10];
  const float* B_bias= (const float*)d_in[11];
  const float* C_bias= (const float*)d_in[12];
  const float* Wout  = (const float*)d_in[13];
  float* out = (float*)d_out;

  char* ws = (char*)d_ws;
  _Float16* xh    = (_Float16*)(ws + o_xh);
  _Float16* Wxh   = (_Float16*)(ws + o_Wxh);
  _Float16* Wzh   = (_Float16*)(ws + o_Wzh);
  _Float16* WBh   = (_Float16*)(ws + o_WBh);
  _Float16* WCh   = (_Float16*)(ws + o_WCh);
  _Float16* Wdth  = (_Float16*)(ws + o_Wdth);
  _Float16* Wlamh = (_Float16*)(ws + o_Wlamh);
  _Float16* Wthh  = (_Float16*)(ws + o_Wthh);
  _Float16* Wouth = (_Float16*)(ws + o_Wouth);
  _Float16* yh    = (_Float16*)(ws + o_yh);
  float* xp    = (float*)(ws + o_xp);
  float* z     = (float*)(ws + o_z);
  float* Bmb   = (float*)(ws + o_Bm);
  float* Cmb   = (float*)(ws + o_Cm);
  float* theta = (float*)(ws + o_th);
  float* dt    = (float*)(ws + o_dt);
  float* lam   = (float*)(ws + o_lam);
  float* al    = (float*)(ws + o_al);
  float* be    = (float*)(ws + o_be);
  float* ga    = (float*)(ws + o_ga);
  float* ybuf  = (float*)(ws + o_y);

  // 1) fp32 -> fp16 (activations + weights; f16 weights ~135MB, L2-resident)
  launch_cvt(x,    xh,    Mrows * Dd,          stream);
  launch_cvt(Wx,   Wxh,   (size_t)DIN * Dd,    stream);
  launch_cvt(Wz,   Wzh,   (size_t)DIN * Dd,    stream);
  launch_cvt(WB,   WBh,   (size_t)NH  * Dd,    stream);
  launch_cvt(WC,   WCh,   (size_t)NH  * Dd,    stream);
  launch_cvt(Wdt,  Wdth,  (size_t)Hh  * Dd,    stream);
  launch_cvt(Wlam, Wlamh, (size_t)Hh  * Dd,    stream);
  launch_cvt(Wth,  Wthh,  (size_t)NH2 * Dd,    stream);
  launch_cvt(Wout, Wouth, (size_t)Dd  * DIN,   stream);

  // 2) projection GEMMs (WMMA f16, fp32 accumulate, async-LDS staged)
  launch_gemm(xh, Wxh,   xp,    Dd, DIN, stream);
  launch_gemm(xh, Wzh,   z,     Dd, DIN, stream);
  launch_gemm(xh, WBh,   Bmb,   Dd, NH,  stream);
  launch_gemm(xh, WCh,   Cmb,   Dd, NH,  stream);
  launch_gemm(xh, Wdth,  dt,    Dd, Hh,  stream);
  launch_gemm(xh, Wlamh, lam,   Dd, Hh,  stream);
  launch_gemm(xh, Wthh,  theta, Dd, NH2, stream);

  // 3) rmsnorm + bias on B and C
  mamba3_rmsnorm_bias_kernel<<<(unsigned)(Mrows * Hh), 128, 0, stream>>>(Bmb, Bn_w, B_bias);
  mamba3_rmsnorm_bias_kernel<<<(unsigned)(Mrows * Hh), 128, 0, stream>>>(Cmb, Cn_w, C_bias);

  // 4) dt/lam -> softplus/sigmoid, alpha/beta/gamma
  mamba3_dtlam_kernel<<<(unsigned)(Mrows * Hh / 256), 256, 0, stream>>>(
      dt, lam, A_log, al, be, ga, (int)(Mrows * Hh));

  // 5) cumsum of theta*dt along L
  mamba3_cumsum_kernel<<<Bb * Hh * 64 / 256, 256, 0, stream>>>(theta, dt);

  // 6) data-dependent RoPE on B and C
  mamba3_rope_kernel<<<(unsigned)(Mrows * Hh * 64 / 256), 256, 0, stream>>>(Bmb, Cmb, theta);

  // 7) selective scan: 128 workgroups (one per (b,h)), state in VGPRs
  mamba3_scan_kernel<<<Bb * Hh, 256, 0, stream>>>(Bmb, Cmb, xp, al, be, ga, ybuf);

  // 8) gate y *= silu(z), convert to f16
  mamba3_gate_cvt_kernel<<<(unsigned)(Mrows * DIN / 256), 256, 0, stream>>>(ybuf, z, yh);

  // 9) output projection: out = y_gated @ Wout^T
  launch_gemm(yh, Wouth, out, DIN, Dd, stream);
}
